// AdaptiveModalitySelectionSystem_21337397527134
// MI455X (gfx1250) — compile-verified, matrix operand order checked by reading, not run on hardware
//
#include <hip/hip_runtime.h>
#include <math.h>

// ---------------------------------------------------------------------------
// Problem constants (match reference)
// ---------------------------------------------------------------------------
#define B_    4096
#define K_    4
#define DIN   1024
#define H_    1024
#define CTX_  256
#define RH    64
#define LN_EPS 1e-5f

typedef __attribute__((ext_vector_type(16))) __bf16        v16bf;
typedef __attribute__((ext_vector_type(8)))  float         v8f;
typedef __attribute__((ext_vector_type(4)))  float         v4f;
typedef __attribute__((ext_vector_type(4)))  unsigned int  v4u;
typedef __attribute__((ext_vector_type(4)))  int           v4i;

// Async global->LDS path (CDNA5): guarded so the file compiles on any toolchain.
#if defined(__has_builtin)
#if __has_builtin(__builtin_amdgcn_global_load_async_to_lds_b128) && \
    __has_builtin(__builtin_amdgcn_s_wait_asynccnt)
#define HAVE_ASYNC 1
#endif
#endif

__device__ __forceinline__ unsigned short f2bf(float f) {
    union { float f; unsigned int u; } v; v.f = f;
    unsigned int u = v.u;
    u += 0x7FFFu + ((u >> 16) & 1u);   // round-to-nearest-even
    return (unsigned short)(u >> 16);
}

__device__ __forceinline__ float sigm(float x) { return 1.0f / (1.0f + __expf(-x)); }

// ---------------------------------------------------------------------------
// Kernel 1: router MLP -> per-(b,k) fused gating coefficient
//   coef[b,k] = hard(b,k) * mask(b,k) * softmax(fusion_w)[k]
// ---------------------------------------------------------------------------
__global__ __launch_bounds__(64) void router_kernel(
    const float* __restrict__ context, const float* __restrict__ gumbel,
    const float* __restrict__ W1, const float* __restrict__ b1,
    const float* __restrict__ g_ln, const float* __restrict__ beta_ln,
    const float* __restrict__ W2, const float* __restrict__ b2,
    const float* __restrict__ W3, const float* __restrict__ b3,
    const float* __restrict__ prior, const float* __restrict__ fusion_w,
    float* __restrict__ coef)
{
    __shared__ float sh[RH];
    __shared__ float sh2[RH];
    __shared__ float sh3[RH / 2];

    const int b = blockIdx.x;
    const int t = threadIdx.x;

    const float* crow = context + (size_t)b * CTX_;
    float acc = b1[t];
    #pragma unroll 4
    for (int d = 0; d < CTX_; ++d) acc += crow[d] * W1[d * RH + t];
    sh[t] = acc;
    __syncthreads();

    float mu = 0.f, var = 0.f;
    for (int j = 0; j < RH; ++j) mu += sh[j];
    mu *= (1.0f / RH);
    for (int j = 0; j < RH; ++j) { float d = sh[j] - mu; var += d * d; }
    var *= (1.0f / RH);
    float hn = (acc - mu) * rsqrtf(var + LN_EPS) * g_ln[t] + beta_ln[t];
    hn = fmaxf(hn, 0.0f);
    __syncthreads();
    sh2[t] = hn;
    __syncthreads();

    if (t < RH / 2) {
        float a2 = b2[t];
        for (int j = 0; j < RH; ++j) a2 += sh2[j] * W2[j * (RH / 2) + t];
        sh3[t] = fmaxf(a2, 0.0f);
    }
    __syncthreads();

    if (t == 0) {
        float probs[K_], soft[K_];
        for (int k = 0; k < K_; ++k) {
            float a3 = b3[k] + prior[k];
            for (int j = 0; j < RH / 2; ++j) a3 += sh3[j] * W3[j * K_ + k];
            probs[k] = sigm(a3);
            soft[k]  = sigm(a3 + gumbel[b * K_ + k]);   // TEMP = 1.0
        }
        int i1 = 0;
        for (int k = 1; k < K_; ++k) if (probs[k] > probs[i1]) i1 = k;
        int i2 = (i1 == 0) ? 1 : 0;
        for (int k = 0; k < K_; ++k) if (k != i1 && probs[k] > probs[i2]) i2 = k;
        float fw[K_];
        float mx = fusion_w[0];
        for (int k = 1; k < K_; ++k) mx = fmaxf(mx, fusion_w[k]);
        float se = 0.f;
        for (int k = 0; k < K_; ++k) { fw[k] = __expf(fusion_w[k] - mx); se += fw[k]; }
        float inv = 1.0f / se;
        for (int k = 0; k < K_; ++k) {
            float mm   = (k == i1 || k == i2) ? 1.0f : 0.0f;
            float mask = fmaxf(soft[k], mm);
            float g    = (mask > 0.5f) ? mask : 0.0f;
            coef[b * K_ + k] = g * fw[k] * inv;
        }
    }
}

// ---------------------------------------------------------------------------
// GEMM geometry
// ---------------------------------------------------------------------------
#define BM 128
#define BN 128
#define BK 32
#define LDST 40                         // A-tile ushort row stride (80B, 16B-aligned)
#define NK  ((K_ * DIN) / BK)           // 128 K-steps
#define BTILE_BYTES (BN * BK * 2)       // 8192B packed bf16 B tile

// A-tile swizzled in-row ushort position for K index c (0..31)
__device__ __forceinline__ int swz(int c) {
    return (((c >> 3) & 1) << 4) + (c & 7) + ((c >> 4) << 3);
}

// Packed B tile: byte offset of element (n 0..127, kk 0..31).
// 512B block per (n-group-of-16, K-half); inside: 256B per fragment quarter,
// 16B per lane row -> per-lane fragment = two contiguous, conflict-free b128.
__device__ __forceinline__ int packOff(int n, int kk) {
    int khalf = (kk >> 3) & 1;
    int pos   = (kk & 7) + ((kk >> 4) << 3);          // 0..15
    int block = ((n >> 4) << 1) + khalf;              // 0..15
    return block * 512 + (pos >> 3) * 256 + (n & 15) * 16 + (pos & 7) * 2;
}

// ---------------------------------------------------------------------------
// Kernel 2: one-time W_enc fp32 -> bf16 pre-swizzled pack (hoists the 32x
// redundant per-block conversion out of the GEMM hot loop).
// bpack layout: [colBlock(8)][kt(128)] tiles of 8192B.
// ---------------------------------------------------------------------------
__global__ __launch_bounds__(256) void pack_W_kernel(
    const float* __restrict__ W_enc, unsigned short* __restrict__ bpack)
{
    int t   = blockIdx.x * 256 + threadIdx.x;   // 0 .. 2^20-1
    int n   = t & 127;
    int kkg = (t >> 7) & 7;
    int kt  = (t >> 10) & 127;
    int cb  = t >> 17;
    int kk0 = kkg * 4;
    int mod = kt >> 5;
    int d0  = (kt & 31) * 32;

    const float* src = W_enc + ((size_t)mod * DIN + d0 + kk0) * H_ + cb * 128 + n;
    unsigned long long pk = 0;
    #pragma unroll
    for (int j = 0; j < 4; ++j)
        pk |= (unsigned long long)f2bf(src[(size_t)j * H_]) << (16 * j);

    unsigned char* dst = (unsigned char*)bpack
                       + (size_t)(cb * NK + kt) * BTILE_BYTES + packOff(n, kk0);
    *(unsigned long long*)dst = pk;
}

// ---------------------------------------------------------------------------
// Kernel 3: fused gated encoder GEMM, bf16 WMMA, double-buffered LDS.
// A: coef-scaled x, converted in-kernel (b128 loads, packed b64 LDS stores).
// B: packed bf16 tiles streamed with async global->LDS copies (ASYNCcnt).
// ---------------------------------------------------------------------------
__global__ __launch_bounds__(256) void fused_enc_kernel(
    const float* __restrict__ x, const unsigned short* __restrict__ bpack,
    const float* __restrict__ b_enc, const float* __restrict__ coef,
    float* __restrict__ out)
{
    __shared__ __align__(16) unsigned short As[2][BM * LDST];
    __shared__ __align__(16) unsigned char  Bs[2][BTILE_BYTES];

    const int tid   = threadIdx.x;
    const int wave  = tid >> 5;
    const int lane  = tid & 31;
    const int waveM = wave & 3;
    const int waveN = wave >> 2;
    const int lrow  = lane & 15;
    const int lhalf = lane >> 4;

    const int blockN = blockIdx.x * BN;       // also the B colBlock
    const int blockM = blockIdx.y * BM;

    // A staging geometry: thread loads float4 at (rA4 + 32*i, cA4)
    const int rA4  = tid >> 3;                // 0..31
    const int cA4  = (tid & 7) * 4;           // 0,4,...,28
    const int swzA = swz(cA4);                // 4 adjacent swizzled positions

    const unsigned char* bpackBytes = (const unsigned char*)bpack
                                    + (size_t)blockIdx.x * NK * BTILE_BYTES;

    union Frag { v16bf v; v4u q[2]; };
    Frag afr[2], bfr[4];

    v8f acc[2][4];
    const v8f vzero = {0.f, 0.f, 0.f, 0.f, 0.f, 0.f, 0.f, 0.f};
    #pragma unroll
    for (int mt = 0; mt < 2; ++mt)
        #pragma unroll
        for (int nt = 0; nt < 4; ++nt) acc[mt][nt] = vzero;

    v4f aReg[4];
    float cf[4];
    int curMod = 0;
#if !defined(HAVE_ASYNC)
    v4u bq0, bq1;
#endif

    #define LOAD_REGS_A(KT)                                                   \
    {                                                                         \
        const int k0_  = (KT) * BK;                                           \
        const int mod_ = k0_ >> 10;                                           \
        const int d0_  = k0_ & (DIN - 1);                                     \
        const float* xa_ = x + ((size_t)mod_ * B_ + blockM) * DIN + d0_;      \
        _Pragma("unroll")                                                     \
        for (int i = 0; i < 4; ++i)                                           \
            aReg[i] = *(const v4f*)(xa_ + (size_t)(rA4 + 32 * i) * DIN + cA4);\
    }

    #define RELOAD_CF(MOD)                                                    \
    {                                                                         \
        _Pragma("unroll")                                                     \
        for (int i = 0; i < 4; ++i)                                           \
            cf[i] = coef[(blockM + rA4 + 32 * i) * K_ + (MOD)];               \
    }

    #define CVT_STORE_A(BUF)                                                  \
    {                                                                         \
        unsigned short* abuf_ = &As[(BUF)][0];                                \
        _Pragma("unroll")                                                     \
        for (int i = 0; i < 4; ++i) {                                         \
            unsigned long long pk_;                                           \
            pk_  = (unsigned long long)f2bf(aReg[i][0] * cf[i]);              \
            pk_ |= (unsigned long long)f2bf(aReg[i][1] * cf[i]) << 16;        \
            pk_ |= (unsigned long long)f2bf(aReg[i][2] * cf[i]) << 32;        \
            pk_ |= (unsigned long long)f2bf(aReg[i][3] * cf[i]) << 48;        \
            *(unsigned long long*)(abuf_ + (rA4 + 32 * i) * LDST + swzA) = pk_;\
        }                                                                     \
    }

#if defined(HAVE_ASYNC)
    // fire-and-forget async copy of the 8KB packed tile (2 x b128 per thread).
    // Builtin signature (from hipcc diagnostics):
    //   (v4i addrspace(1)* src, v4i addrspace(3)* dst, imm offset, imm cpol)
    #define ISSUE_B(KT, BUF)                                                  \
    {                                                                         \
        const unsigned char* s_ = bpackBytes + (size_t)(KT) * BTILE_BYTES     \
                                             + tid * 32;                      \
        unsigned char* d_ = &Bs[(BUF)][tid * 32];                             \
        auto gp0_ = (__attribute__((address_space(1))) v4i*)                  \
                        (unsigned long long)(s_);                             \
        auto gp1_ = (__attribute__((address_space(1))) v4i*)                  \
                        (unsigned long long)(s_ + 16);                        \
        auto lp0_ = (__attribute__((address_space(3))) v4i*)                  \
                        (unsigned int)(unsigned long long)(d_);               \
        auto lp1_ = (__attribute__((address_space(3))) v4i*)                  \
                        (unsigned int)(unsigned long long)(d_ + 16);          \
        __builtin_amdgcn_global_load_async_to_lds_b128(gp0_, lp0_, 0, 0);     \
        __builtin_amdgcn_global_load_async_to_lds_b128(gp1_, lp1_, 0, 0);     \
    }
    #define COMMIT_B(BUF)  { }
    #define ASYNC_WAIT()   { __builtin_amdgcn_s_wait_asynccnt(0); }
#else
    #define ISSUE_B(KT, BUF)                                                  \
    {                                                                         \
        const v4u* s_ = (const v4u*)(bpackBytes + (size_t)(KT) * BTILE_BYTES) \
                        + tid * 2;                                            \
        bq0 = s_[0]; bq1 = s_[1];                                             \
    }
    #define COMMIT_B(BUF)                                                     \
    {                                                                         \
        *(v4u*)&Bs[(BUF)][tid * 32]      = bq0;                               \
        *(v4u*)&Bs[(BUF)][tid * 32 + 16] = bq1;                               \
    }
    #define ASYNC_WAIT()   { }
#endif

    // ---------------- prologue: fill buffer 0 ----------------
    LOAD_REGS_A(0);
    ISSUE_B(0, 0);
    RELOAD_CF(0);
    CVT_STORE_A(0);
    COMMIT_B(0);
    ASYNC_WAIT();
    __syncthreads();

    int cur = 0;
    for (int kt = 0; kt < NK; ++kt) {
        const bool more = (kt + 1 < NK);
        if (more) {
            LOAD_REGS_A(kt + 1);          // global loads in flight over the WMMAs
            ISSUE_B(kt + 1, cur ^ 1);     // async engine streams B tile
        }
        if (kt + 2 < NK) {                // warm L2 two tiles ahead
            int k2 = (kt + 2) * BK;
            int mod2 = k2 >> 10, d2 = k2 & (DIN - 1);
            __builtin_prefetch(x + ((size_t)mod2 * B_ + blockM + rA4) * DIN + d2, 0, 0);
            __builtin_prefetch(bpackBytes + (size_t)(kt + 2) * BTILE_BYTES + tid * 32, 0, 0);
        }

        // ---- fragment gathers: two ds_load_b128 each ----
        const unsigned short* ab = &As[cur][0];
        const unsigned char*  bb = &Bs[cur][0];
        #pragma unroll
        for (int mt = 0; mt < 2; ++mt) {
            int m = waveM * 32 + mt * 16 + lrow;
            const v4u* p = (const v4u*)(ab + m * LDST + lhalf * 16);
            afr[mt].q[0] = p[0];
            afr[mt].q[1] = p[1];
        }
        #pragma unroll
        for (int nt = 0; nt < 4; ++nt) {
            const unsigned char* p = bb + (((waveN * 4 + nt) * 2 + lhalf) << 9)
                                        + (lrow << 4);
            bfr[nt].q[0] = *(const v4u*)(p);
            bfr[nt].q[1] = *(const v4u*)(p + 256);
        }

        // ---- 8x v_wmma_f32_16x16x32_bf16 ----
        #pragma unroll
        for (int mt = 0; mt < 2; ++mt)
            #pragma unroll
            for (int nt = 0; nt < 4; ++nt)
                acc[mt][nt] = __builtin_amdgcn_wmma_f32_16x16x32_bf16(
                    false, afr[mt].v, false, bfr[nt].v,
                    (short)0, acc[mt][nt], false, false);

        if (more) {
            int m1 = (kt + 1) >> 5;
            if (m1 != curMod) { curMod = m1; RELOAD_CF(m1); }
            CVT_STORE_A(cur ^ 1);
            COMMIT_B(cur ^ 1);
            ASYNC_WAIT();
        }
        __syncthreads();
        cur ^= 1;
    }

    // ---- epilogue: add folded bias  sum_k coef[b,k]*b_enc[k,h] ----
    #pragma unroll
    for (int mt = 0; mt < 2; ++mt) {
        #pragma unroll
        for (int nt = 0; nt < 4; ++nt) {
            int hcol = blockN + waveN * 64 + nt * 16 + lrow;
            #pragma unroll
            for (int r = 0; r < 8; ++r) {
                int m = blockM + waveM * 32 + mt * 16 + lhalf * 8 + r;
                float bias = 0.f;
                #pragma unroll
                for (int k = 0; k < K_; ++k)
                    bias += coef[m * K_ + k] * b_enc[k * H_ + hcol];
                out[(size_t)m * H_ + hcol] = acc[mt][nt][r] + bias;
            }
        }
    }
    #undef LOAD_REGS_A
    #undef RELOAD_CF
    #undef CVT_STORE_A
    #undef ISSUE_B
    #undef COMMIT_B
    #undef ASYNC_WAIT
}

// ---------------------------------------------------------------------------
// Host launcher
// ---------------------------------------------------------------------------
extern "C" void kernel_launch(void* const* d_in, const int* in_sizes, int n_in,
                              void* d_out, int out_size, void* d_ws, size_t ws_size,
                              hipStream_t stream) {
    (void)in_sizes; (void)n_in; (void)out_size; (void)ws_size;
    const float* context  = (const float*)d_in[0];
    const float* x        = (const float*)d_in[1];
    const float* gumbel   = (const float*)d_in[2];
    const float* W1       = (const float*)d_in[3];
    const float* b1       = (const float*)d_in[4];
    const float* g_ln     = (const float*)d_in[5];
    const float* beta_ln  = (const float*)d_in[6];
    const float* W2       = (const float*)d_in[7];
    const float* b2       = (const float*)d_in[8];
    const float* W3       = (const float*)d_in[9];
    const float* b3       = (const float*)d_in[10];
    const float* prior    = (const float*)d_in[11];
    const float* W_enc    = (const float*)d_in[12];
    const float* b_enc    = (const float*)d_in[13];
    const float* fusion_w = (const float*)d_in[14];

    float*          coef  = (float*)d_ws;                       // 64 KB
    unsigned short* bpack = (unsigned short*)((char*)d_ws + 65536);  // 8 MB

    float* out = (float*)d_out;

    router_kernel<<<B_, 64, 0, stream>>>(context, gumbel, W1, b1, g_ln, beta_ln,
                                         W2, b2, W3, b3, prior, fusion_w, coef);

    pack_W_kernel<<<4096, 256, 0, stream>>>(W_enc, bpack);

    dim3 grid(H_ / BN, B_ / BM);         // 8 x 32 = 256 workgroups, 8 waves each
    fused_enc_kernel<<<grid, 256, 0, stream>>>(x, bpack, b_enc, coef, out);
}
